// DeformConv_69252052681176
// MI455X (gfx1250) — matrix operand order repaired and checked by reading, not compiled, not run on hardware
//
#include <hip/hip_runtime.h>
#include <hip/hip_bf16.h>
#include <math.h>

#define N_    4
#define CIN   64
#define COUT  64
#define H_    128
#define W_    128
#define HW    (H_ * W_)
#define KTAPS 9
#define COFF  27

typedef __attribute__((ext_vector_type(16))) __bf16 v16bf;
typedef __attribute__((ext_vector_type(8)))  __bf16 v8bf;
typedef __attribute__((ext_vector_type(8)))  float  v8f;
typedef __attribute__((ext_vector_type(2)))  float  v2f;
typedef __attribute__((ext_vector_type(4)))  unsigned int u32x4;
typedef __attribute__((ext_vector_type(8)))  int          i32x8;
typedef __attribute__((ext_vector_type(4)))  int          i32x4;

// ---------------------------------------------------------------------------
// Kernel 1: offset conv (3x3, Cin=64 -> 27ch) as f32 WMMA GEMM.
// M = positions (64/block, one half-row), N = 32 cols (27 valid), K = 576.
// A tile kept K-major (Acs[ci][pos]) = exactly the layout the Tensor Data
// Mover produces for a 2D (64 x 64-DWORD-line) tile, so interior taps are
// staged by TDM (tensor_load_to_lds, TENSORcnt) and only image-boundary taps
// use the zero-padding VALU path.
// ---------------------------------------------------------------------------
__global__ __launch_bounds__(256) void offset_conv_wmma(
    const float* __restrict__ xin,
    const float* __restrict__ w_off,
    const float* __restrict__ b_off,
    float* __restrict__ off)
{
  __shared__ float Acs[CIN][64];       // K-major im2col chunk (ci, pos)
  __shared__ float BsT[32][CIN + 4];   // transposed weight slice

  const int tid  = threadIdx.x;
  const int lane = tid & 31;
  const int wv   = tid >> 5;          // wave 0..7
  const int pm   = wv >> 1;           // position subtile 0..3
  const int pn   = wv & 1;            // col subtile 0..1
  const int p0   = blockIdx.x * 64;

  // Block-uniform tile coordinates (64 positions lie in one image row)
  const int bn = p0 / HW;
  const int bh = (p0 - bn * HW) / W_;
  const int bw = p0 & (W_ - 1);

  const int rowA = pm * 16 + (lane & 15);
  const int colB = pn * 16 + (lane & 15);
  const int koff = (lane >> 4) * 2;   // f32 A 16x4: lanes 0-15 K0/K1, 16-31 K2/K3

  v8f acc = {};

  for (int tap = 0; tap < KTAPS; ++tap) {
    const int ky = tap / 3, kx = tap % 3;
    const int y  = bh + ky - 1;
    const int x0 = bw + kx - 1;
    const bool interior = (y >= 0) && (y < H_) && (x0 >= 0) && (x0 + 64 <= W_);

    if (interior) {
      // ---- TDM path: one 2D tile DMA, global -> LDS (K-major layout) ----
      if (wv == 0) {
        unsigned long long ga =
            (unsigned long long)(uintptr_t)&xin[((bn * CIN) * H_ + y) * W_ + x0];
        unsigned int lds = (unsigned int)(uintptr_t)&Acs[0][0];

        u32x4 g0;
        g0[0] = 1u;                                   // count=1, user descriptor
        g0[1] = lds;                                  // lds_addr
        g0[2] = (unsigned int)ga;                     // global_addr[31:0]
        g0[3] = (unsigned int)((ga >> 32) & 0x1FFFFFFu) | (2u << 30); // type=2

        const unsigned int TD0 = 1u << 20;            // huge dims: no OOB clip
        const unsigned int TD1 = 1u << 20;
        i32x8 g1;
        g1[0] = (int)0x00020000u;                     // data_size=2 (4 bytes)
        g1[1] = (int)((TD0 & 0xFFFFu) << 16);         // tensor_dim0 lo16
        g1[2] = (int)(((TD0 >> 16) & 0xFFFFu) |
                      ((TD1 & 0xFFFFu) << 16));       // dim0 hi16 | dim1 lo16
        g1[3] = (int)(((TD1 >> 16) & 0xFFFFu) |
                      (64u << 16));                   // dim1 hi16 | tile_dim0=64
        g1[4] = 64;                                   // tile_dim1=64, tile_dim2=0
        g1[5] = HW;                                   // tensor_dim0_stride = H*W
        g1[6] = 0;
        g1[7] = 0;

        i32x4 gz4 = {0, 0, 0, 0};
        i32x8 gz8 = {0, 0, 0, 0, 0, 0, 0, 0};
        __builtin_amdgcn_tensor_load_to_lds(g0, g1, gz4, gz4, gz8, 0);
        __builtin_amdgcn_s_wait_tensorcnt(0);
      }
    } else {
      // ---- VALU path with zero padding (image border tiles) ----
      for (int j = 0; j < 16; ++j) {
        int idx = tid + 256 * j;
        int ci  = idx >> 6;
        int i   = idx & 63;
        int xx  = x0 + i;
        float v = 0.f;
        if (y >= 0 && y < H_ && xx >= 0 && xx < W_)
          v = xin[((bn * CIN + ci) * H_ + y) * W_ + xx];
        Acs[ci][i] = v;
      }
    }

    // Stage B transposed: BsT[co][ci] = w_off[co][ci][ky][kx] (co 27..31 = 0)
    for (int j = 0; j < 8; ++j) {
      int idx = tid + 256 * j;
      int ci  = idx & 63;
      int co  = idx >> 6;              // 0..31
      float v = 0.f;
      if (co < COFF)
        v = w_off[((co * CIN + ci) * 3 + ky) * 3 + kx];
      BsT[co][ci] = v;
    }
    __syncthreads();

    // 16 x V_WMMA_F32_16X16X4_F32 over this tap's K=64
    for (int kc = 0; kc < 16; ++kc) {
      int k0 = kc * 4 + koff;
      v2f av;
      av.x = Acs[k0    ][rowA];        // lane-consecutive ds_load_b32
      av.y = Acs[k0 + 1][rowA];
      v2f bv = *(const v2f*)&BsT[colB][k0];   // ds_load_b64
      acc = __builtin_amdgcn_wmma_f32_16x16x4_f32(
          false, av, false, bv, (short)0, acc, false, false);
    }
    __syncthreads();
  }

  // Epilogue: + b_off, store to offsets workspace (NCHW, C=27)
  if (colB < COFF) {
    const float bias = b_off[colB];
    for (int r = 0; r < 8; ++r) {
      int i = pm * 16 + r + ((lane >> 4) << 3);   // C layout: lanes16-31 -> M+8
      off[((bn * COFF + colB) * H_ + bh) * W_ + (bw + i)] = acc[r] + bias;
    }
  }
}

// ---------------------------------------------------------------------------
// Kernel 2: fused modulated-bilinear sampling + bf16 WMMA GEMM + BN + ReLU.
// M = positions (32/block), N = 64 Cout, K = 576 in 9 taps of 64.
// 8 waves per block = 2 pos-tiles x 4 cout-tiles of 16x16.
// A and transposed B staged as bf16 in LDS; all fragment traffic is b128.
// ---------------------------------------------------------------------------
__global__ __launch_bounds__(256) void dcn_main_wmma(
    const float* __restrict__ xin,
    const float* __restrict__ off,
    const float* __restrict__ w_dcn,
    const float* __restrict__ b_dcn,
    const float* __restrict__ bn_g,
    const float* __restrict__ bn_b,
    const float* __restrict__ bn_m,
    const float* __restrict__ bn_v,
    float* __restrict__ out)
{
  __shared__ __bf16 As [32][CIN + 8];    // sampled chunk, row stride 144 B
  __shared__ __bf16 BsT[COUT][CIN + 8];  // transposed weight slice

  const int tid  = threadIdx.x;
  const int lane = tid & 31;
  const int wv   = tid >> 5;
  const int pm   = wv >> 2;             // position subtile 0..1
  const int pn   = wv & 3;              // cout subtile 0..3
  const int p0   = blockIdx.x * 32;

  const int rowA  = pm * 16 + (lane & 15);
  const int nn    = pn * 16 + (lane & 15);
  const int koffA = (lane >> 4) * 8;    // 16-bit A 16x32 lane split
  const int koffB = (lane >> 4) * 8;    // 16-bit B 32x16 lane split

  // Sampling work split: thread -> (position, 8-channel group)
  const int si   = tid >> 3;            // 0..31
  const int cg   = tid & 7;
  const int sp   = p0 + si;
  const int sn   = sp / HW;
  const int srem = sp - sn * HW;
  const int sh   = srem / W_;
  const int sw   = srem % W_;

  v8f acc = {};

  for (int tap = 0; tap < KTAPS; ++tap) {
    const int ky = tap / 3, kx = tap % 3;
    if (tap + 1 < KTAPS)
      __builtin_prefetch(&w_dcn[tap + 1], 0, 1);   // global_prefetch_b8

    // Stage B transposed: BsT[co][ci] = bf16(w_dcn[co][ci][tap])
    for (int j = 0; j < 16; ++j) {
      int idx = tid + 256 * j;
      int ci  = idx & 63;
      int co  = idx >> 6;
      BsT[co][ci] = (__bf16)w_dcn[(co * CIN + ci) * KTAPS + tap];
    }

    // Stage A: modulated bilinear sample of 32 pos x 64 ci
    {
      float o1 = off[((sn * COFF + tap)             * H_ + sh) * W_ + sw];
      float o2 = off[((sn * COFF + KTAPS + tap)     * H_ + sh) * W_ + sw];
      float mm = off[((sn * COFF + 2 * KTAPS + tap) * H_ + sh) * W_ + sw];
      float msk = 1.f / (1.f + __expf(-mm));
      float fy  = o1 + (float)(ky + sh - 1);
      float fx  = o2 + (float)(kx + sw - 1);
      float fy0 = floorf(fy), fx0 = floorf(fx);
      float wy  = fy - fy0,   wx  = fx - fx0;
      int y0 = (int)fy0, x0 = (int)fx0;
      int y1 = y0 + 1,   x1 = x0 + 1;
      bool iy0 = (y0 >= 0) && (y0 < H_);
      bool iy1 = (y1 >= 0) && (y1 < H_);
      bool ix0 = (x0 >= 0) && (x0 < W_);
      bool ix1 = (x1 >= 0) && (x1 < W_);
      int yc0 = y0 < 0 ? 0 : (y0 > H_ - 1 ? H_ - 1 : y0);
      int yc1 = y1 < 0 ? 0 : (y1 > H_ - 1 ? H_ - 1 : y1);
      int xc0 = x0 < 0 ? 0 : (x0 > W_ - 1 ? W_ - 1 : x0);
      int xc1 = x1 < 0 ? 0 : (x1 > W_ - 1 ? W_ - 1 : x1);
      float w00 = (1.f - wy) * (1.f - wx) * msk * (float)(iy0 && ix0);
      float w01 = (1.f - wy) * wx         * msk * (float)(iy0 && ix1);
      float w10 = wy         * (1.f - wx) * msk * (float)(iy1 && ix0);
      float w11 = wy         * wx         * msk * (float)(iy1 && ix1);
      const float* base = xin + sn * CIN * HW;
      v8bf pack;
      for (int c = 0; c < 8; ++c) {
        int ci = cg * 8 + c;
        const float* ip = base + ci * HW;
        float s = ip[yc0 * W_ + xc0] * w00
                + ip[yc0 * W_ + xc1] * w01
                + ip[yc1 * W_ + xc0] * w10
                + ip[yc1 * W_ + xc1] * w11;
        pack[c] = (__bf16)s;
      }
      *(v8bf*)&As[si][cg * 8] = pack;            // one ds_store_b128
    }
    __syncthreads();

    // 2 x V_WMMA_F32_16X16X32_BF16 over this tap's K=64
    for (int half = 0; half < 2; ++half) {
      int kb = half * 32;
      v8bf alo = *(const v8bf*)&As [rowA][kb + koffA];        // ds_load_b128
      v8bf ahi = *(const v8bf*)&As [rowA][kb + 16 + koffA];   // ds_load_b128
      v8bf blo = *(const v8bf*)&BsT[nn  ][kb + koffB];        // ds_load_b128
      v8bf bhi = *(const v8bf*)&BsT[nn  ][kb + koffB + 16];   // ds_load_b128
      v16bf a = __builtin_shufflevector(alo, ahi, 0, 1, 2, 3, 4, 5, 6, 7,
                                        8, 9, 10, 11, 12, 13, 14, 15);
      v16bf b = __builtin_shufflevector(blo, bhi, 0, 1, 2, 3, 4, 5, 6, 7,
                                        8, 9, 10, 11, 12, 13, 14, 15);
      acc = __builtin_amdgcn_wmma_f32_16x16x32_bf16(
          false, a, false, b, (short)0, acc, false, false);
    }
    __syncthreads();
  }

  // Epilogue: conv bias + BatchNorm + ReLU, store NCHW f32
  const float scale = bn_g[nn] * rsqrtf(bn_v[nn] + 1e-5f);
  const float shift = bn_b[nn];
  const float bias  = b_dcn[nn];
  const float mean  = bn_m[nn];
  for (int r = 0; r < 8; ++r) {
    int i = pm * 16 + r + ((lane >> 4) << 3);
    int p = p0 + i;
    int n = p / HW;
    int rem = p - n * HW;
    int h = rem / W_;
    int w = rem % W_;
    float v = (acc[r] + bias - mean) * scale + shift;
    out[((n * COUT + nn) * H_ + h) * W_ + w] = fmaxf(v, 0.f);
  }
}

// ---------------------------------------------------------------------------
extern "C" void kernel_launch(void* const* d_in, const int* in_sizes, int n_in,
                              void* d_out, int out_size, void* d_ws, size_t ws_size,
                              hipStream_t stream) {
  const float* xin   = (const float*)d_in[0];
  const float* w_off = (const float*)d_in[1];
  const float* b_off = (const float*)d_in[2];
  const float* w_dcn = (const float*)d_in[3];
  const float* b_dcn = (const float*)d_in[4];
  const float* bn_g  = (const float*)d_in[5];
  const float* bn_b  = (const float*)d_in[6];
  const float* bn_m  = (const float*)d_in[7];
  const float* bn_v  = (const float*)d_in[8];
  (void)in_sizes; (void)n_in; (void)out_size; (void)ws_size;

  float* off = (float*)d_ws;      // N*27*H*W f32 = ~6.75 MB scratch
  float* out = (float*)d_out;

  offset_conv_wmma<<<(N_ * HW) / 64, 256, 0, stream>>>(xin, w_off, b_off, off);
  dcn_main_wmma<<<(N_ * HW) / 32, 256, 0, stream>>>(xin, off, w_dcn, b_dcn,
                                                    bn_g, bn_b, bn_m, bn_v, out);
}
